// HSTUBlock_40991167873564
// MI455X (gfx1250) — compile-verified, matrix operand order checked by reading, not run on hardware
//
#include <hip/hip_runtime.h>

// ---------------------------------------------------------------------------
// HSTU block for MI455X (gfx1250): bf16 WMMA GEMMs + fused flash-style
// silu-attention with relative position bias, f32 accumulation throughout.
// All row-major tile staging done by the Tensor Data Mover
// (tensor_load_to_lds) with hardware LDS padding; GEMMs double-buffer
// against TENSORcnt. B=2, S=2048, D=1024, H=16, HD=64, rows R = B*S = 4096.
// ---------------------------------------------------------------------------

typedef __attribute__((ext_vector_type(16))) __bf16 v16bf;
typedef __attribute__((ext_vector_type(8)))  __bf16 bf16x8;
typedef __attribute__((ext_vector_type(8)))  float  v8f;
typedef __attribute__((ext_vector_type(4)))  unsigned int u32x4;
typedef __attribute__((ext_vector_type(8)))  int i32x8;
typedef __attribute__((ext_vector_type(4)))  int i32x4;

#define N_B   2
#define N_S   2048
#define N_D   1024
#define N_H   16
#define N_HD  64
#define N_R   4096      // B*S
#define N_4D  4096
#define POS_C 4095      // M-1

#if defined(__has_builtin)
#  if __has_builtin(__builtin_amdgcn_tensor_load_to_lds)
#    define HAVE_TDM 1
#  endif
#endif
#ifndef HAVE_TDM
#  define HAVE_TDM 0
#endif

// ---- helpers --------------------------------------------------------------

static __device__ inline __bf16 to_bf16(float f) {
  unsigned u = __builtin_bit_cast(unsigned, f);
  unsigned r = u + 0x7FFFu + ((u >> 16) & 1u);   // round-to-nearest-even
  unsigned short s = (unsigned short)(r >> 16);
  return __builtin_bit_cast(__bf16, s);
}

static __device__ inline float silu(float c) {
  return c / (1.0f + __expf(-c));
}

struct FragPair { bf16x8 lo, hi; };

// Load one 16x32 bf16 WMMA A-fragment (or B^T-fragment) from LDS.
// Lane L (0..15) holds row (row+L), K = {kb..kb+7, 16+kb..16+kb+7} with
// kb = 0 for lanes 0..15 and 8 for lanes 16..31 (CDNA5 16-bit A layout).
static __device__ inline v16bf load_frag(const __bf16* base, int row, int ld, int kofs) {
  int lane = threadIdx.x & 31;
  int r  = row + (lane & 15);
  int kb = (lane & 16) ? 8 : 0;
  const __bf16* p = base + r * ld + kofs + kb;
  FragPair fp;
  fp.lo = *(const bf16x8*)p;          // ds_load_b128
  fp.hi = *(const bf16x8*)(p + 16);   // ds_load_b128
  return __builtin_bit_cast(v16bf, fp);
}

#if HAVE_TDM
// Issue one TDM 2-D tile load: global (row-major, row_stride elems of 2B)
// -> LDS at lds_off, tile_w x tile_h bf16, with LDS padding so each tile
// row lands on a padded stride (pad_amt DWORDs after every pad_int DWORDs).
// D# layout per CDNA5 ISA (group0 128b, group1 256b).
static __device__ inline void tdm_load_2d(const void* gptr, unsigned lds_off,
                                          unsigned tensor_w, unsigned tensor_h,
                                          unsigned row_stride,
                                          unsigned tile_w, unsigned tile_h,
                                          unsigned pad_int_c, unsigned pad_amt_c) {
  unsigned long long ga = (unsigned long long)gptr;
  u32x4 g0;
  g0[0] = 1u;                                        // count=1, user descriptor
  g0[1] = lds_off;                                   // LDS byte address
  g0[2] = (unsigned)(ga & 0xFFFFFFFFu);              // global_addr[31:0]
  g0[3] = (unsigned)((ga >> 32) & 0x01FFFFFFu)       // global_addr[56:32]
        | (2u << 30);                                // type=2 ("image")
  i32x8 g1;
  g1[0] = (int)((1u << 16)                           // data_size=1 -> 2 bytes
              | (1u << 20)                           // pad_enable
              | (pad_int_c << 22)                    // pad_interval code
              | (pad_amt_c << 25));                  // pad_amount code
  g1[1] = (int)((tensor_w & 0xFFFFu) << 16);         // tensor_dim0[15:0]
  g1[2] = (int)(((tensor_w >> 16) & 0xFFFFu)         // tensor_dim0[31:16]
              | ((tensor_h & 0xFFFFu) << 16));       // tensor_dim1[15:0]
  g1[3] = (int)(((tensor_h >> 16) & 0xFFFFu)         // tensor_dim1[31:16]
              | ((tile_w & 0xFFFFu) << 16));         // tile_dim0
  g1[4] = (int)(tile_h & 0xFFFFu);                   // tile_dim1 (tile_dim2=0)
  g1[5] = (int)row_stride;                           // tensor_dim0_stride[31:0]
  g1[6] = 0;                                         // stride[47:32], dim1_stride lo
  g1[7] = 0;
  i32x4 z4 = {0, 0, 0, 0};
#if __clang_major__ >= 23
  i32x8 z8 = {0, 0, 0, 0, 0, 0, 0, 0};
  __builtin_amdgcn_tensor_load_to_lds(g0, g1, z4, z4, z8, 0);
#else
  __builtin_amdgcn_tensor_load_to_lds(g0, g1, z4, z4, 0);
#endif
}
#endif  // HAVE_TDM

// ---- kernel 0: f32 -> bf16 convert ---------------------------------------

__global__ __launch_bounds__(256) void cvt_bf16_kernel(const float* __restrict__ in,
                                                       __bf16* __restrict__ out, int n) {
  int i = blockIdx.x * 256 + threadIdx.x;
  if (i < n) out[i] = to_bf16(in[i]);
}

// ---- kernel 1: h = silu(x @ w1^T + b1) -> u(f32), q,k,v(bf16) -------------
// C[m,n] = sum_d X[m,d] * W1[n,d];  M = 4096, N = 4096, K = 1024.
// Block tile 128x128x32, 8 waves, wave tile 64x32 (4x2 WMMA tiles).
// Tiles staged by TDM (double-buffered) with hardware LDS padding:
// row = 32 bf16 = 16 DWORDs (interval code 3), pad 4 DWORDs (code 3) -> 40.

#define G_LDS 40                  // LDS row stride (bf16): 64B row + 16B pad
#define TILE_ELEMS (128 * G_LDS)  // one 128-row tile buffer
#define TILE_BYTES (TILE_ELEMS * 2)

__global__ __launch_bounds__(256) void gemm1_kernel(
    const __bf16* __restrict__ A,     // xb  [4096,1024]
    const __bf16* __restrict__ Bw,    // w1b [4096,1024]
    const float*  __restrict__ bias1, // [4096]
    float*  __restrict__ u,
    __bf16* __restrict__ qb, __bf16* __restrict__ kb, __bf16* __restrict__ vb) {
#if HAVE_TDM
  __shared__ __bf16 As[2 * TILE_ELEMS];
  __shared__ __bf16 Bs[2 * TILE_ELEMS];
#else
  __shared__ __bf16 As[TILE_ELEMS];
  __shared__ __bf16 Bs[TILE_ELEMS];
#endif
  const int K = N_D;
  const int m0 = blockIdx.y * 128;
  const int n0 = blockIdx.x * 128;
  const int t = threadIdx.x;
  const int lane = t & 31, w = t >> 5;
  const int wm = (w >> 2) * 64;   // 2 wave rows
  const int wn = (w & 3) * 32;    // 4 wave cols

  v8f acc[4][2];
#pragma unroll
  for (int i = 0; i < 4; ++i)
#pragma unroll
    for (int j = 0; j < 2; ++j) acc[i][j] = (v8f)(0.0f);

#if HAVE_TDM
  const unsigned lds_a = (unsigned)(unsigned long long)(size_t)(void*)As;
  const unsigned lds_b = (unsigned)(unsigned long long)(size_t)(void*)Bs;
  if (w == 0) {  // preload k0 = 0 into buffer 0 (TDM ignores EXEC; 1 issue/wave)
    tdm_load_2d(A  + (size_t)m0 * K, lds_a, K, N_R,  K, 32, 128, 3, 3);
    tdm_load_2d(Bw + (size_t)n0 * K, lds_b, K, N_4D, K, 32, 128, 3, 3);
  }
  for (int k0 = 0; k0 < K; k0 += 32) {
    const int cur = (k0 >> 5) & 1;
    const __bf16* Ac = As + cur * TILE_ELEMS;
    const __bf16* Bc = Bs + cur * TILE_ELEMS;
    if (w == 0) {
      if (k0 + 32 < K) {  // prefetch next tile into the other buffer
        const int nb = cur ^ 1;
        tdm_load_2d(A  + (size_t)m0 * K + (k0 + 32), lds_a + nb * TILE_BYTES,
                    K, N_R,  K, 32, 128, 3, 3);
        tdm_load_2d(Bw + (size_t)n0 * K + (k0 + 32), lds_b + nb * TILE_BYTES,
                    K, N_4D, K, 32, 128, 3, 3);
        __builtin_amdgcn_s_wait_tensorcnt(2);  // current pair has landed
      } else {
        __builtin_amdgcn_s_wait_tensorcnt(0);
      }
    }
    __syncthreads();  // publish current tiles to all waves

    v16bf af[4], bfr[2];
#pragma unroll
    for (int i = 0; i < 4; ++i) af[i]  = load_frag(Ac, wm + i * 16, G_LDS, 0);
#pragma unroll
    for (int j = 0; j < 2; ++j) bfr[j] = load_frag(Bc, wn + j * 16, G_LDS, 0);
#pragma unroll
    for (int i = 0; i < 4; ++i)
#pragma unroll
      for (int j = 0; j < 2; ++j)
        acc[i][j] = __builtin_amdgcn_wmma_f32_16x16x32_bf16(
            false, af[i], false, bfr[j], (short)0, acc[i][j], false, false);
    __syncthreads();  // all waves done before TDM overwrites this buffer
  }
#else
  for (int k0 = 0; k0 < K; k0 += 32) {
#pragma unroll
    for (int i = 0; i < 2; ++i) {
      int c = t + 256 * i;
      int r = c >> 2, c8 = (c & 3) * 8;
      *(bf16x8*)&As[r * G_LDS + c8] = *(const bf16x8*)&A [(size_t)(m0 + r) * K + k0 + c8];
      *(bf16x8*)&Bs[r * G_LDS + c8] = *(const bf16x8*)&Bw[(size_t)(n0 + r) * K + k0 + c8];
    }
    __syncthreads();
    v16bf af[4], bfr[2];
#pragma unroll
    for (int i = 0; i < 4; ++i) af[i]  = load_frag(As, wm + i * 16, G_LDS, 0);
#pragma unroll
    for (int j = 0; j < 2; ++j) bfr[j] = load_frag(Bs, wn + j * 16, G_LDS, 0);
#pragma unroll
    for (int i = 0; i < 4; ++i)
#pragma unroll
      for (int j = 0; j < 2; ++j)
        acc[i][j] = __builtin_amdgcn_wmma_f32_16x16x32_bf16(
            false, af[i], false, bfr[j], (short)0, acc[i][j], false, false);
    __syncthreads();
  }
#endif

  const int cn = lane & 15, rm0 = (lane >> 4) * 8;
#pragma unroll
  for (int i = 0; i < 4; ++i)
#pragma unroll
    for (int j = 0; j < 2; ++j) {
      int n = n0 + wn + j * 16 + cn;
      int seg = n >> 10, nn = n & 1023;
      float bv = bias1[n];
#pragma unroll
      for (int p = 0; p < 8; ++p) {
        int m = m0 + wm + i * 16 + rm0 + p;
        float hv = silu(acc[i][j][p] + bv);
        size_t idx = (size_t)m * N_D + nn;
        if      (seg == 0) u [idx] = hv;
        else if (seg == 1) qb[idx] = to_bf16(hv);
        else if (seg == 2) kb[idx] = to_bf16(hv);
        else               vb[idx] = to_bf16(hv);
      }
    }
}

// ---- kernel 2: fused causal silu-attention --------------------------------
// Per (b,h), 64-query tile: stream 32-key blocks,
//   s = silu(q k^T + pos_w[M-1 + j - i]) * (j<=i);  y += s @ v;  y *= u.
// q (64x64, once) and k (32x64, per block) staged by TDM with padded rows:
// row = 64 bf16 = 32 DWORDs (interval code 4), pad 4 DWORDs (code 3) -> 72.
// v passes through VGPRs for the LDS transpose (TDM cannot transpose).

__global__ __launch_bounds__(256) void attn_kernel(
    const __bf16* __restrict__ qb, const __bf16* __restrict__ kb,
    const __bf16* __restrict__ vb, const float* __restrict__ u,
    const float* __restrict__ pos_w, float* __restrict__ y) {
  __shared__ __bf16 q_s [64 * 72];
  __shared__ __bf16 k_s [32 * 72];
  __shared__ __bf16 s_s [64 * G_LDS];
  __shared__ __bf16 vt_s[64 * G_LDS];

  const int i0 = blockIdx.x * 64;
  const int bh = blockIdx.y;
  const int b = bh >> 4, hh = bh & 15;
  const size_t hoff = (size_t)b * N_S * N_D + hh * N_HD;
  const __bf16* qp = qb + hoff;
  const __bf16* kp = kb + hoff;
  const __bf16* vp = vb + hoff;

  const int t = threadIdx.x, lane = t & 31, w = t >> 5;
  const int cn = lane & 15, rm0 = (lane >> 4) * 8;
  const int wmS = (w >> 1) * 16, wnS = (w & 1) * 16;  // 4x2 s-tiles (64x32)
  const int wmY = (w >> 1) * 16, wnY = (w & 1) * 32;  // 4x2 y-tiles (64x64)

#if HAVE_TDM
  const unsigned lds_q = (unsigned)(unsigned long long)(size_t)(void*)q_s;
  const unsigned lds_k = (unsigned)(unsigned long long)(size_t)(void*)k_s;
  if (w == 0)   // stage the 64x64 query tile once via TDM
    tdm_load_2d(qp + (size_t)i0 * N_D, lds_q, N_D, N_S, N_D, 64, 64, 4, 3);
#else
#pragma unroll
  for (int i = 0; i < 2; ++i) {
    int c = t + 256 * i;
    int r = c >> 3, c8 = (c & 7) * 8;
    *(bf16x8*)&q_s[r * 72 + c8] = *(const bf16x8*)&qp[(size_t)(i0 + r) * N_D + c8];
  }
#endif

  v8f accy0 = (v8f)(0.0f), accy1 = (v8f)(0.0f);
  const int nkb = (i0 >> 5) + 2;             // causal: keys up to i0+63

  for (int kbi = 0; kbi < nkb; ++kbi) {
    const int j0 = kbi * 32;
#if HAVE_TDM
    if (w == 0)  // 32x64 key tile via TDM
      tdm_load_2d(kp + (size_t)j0 * N_D, lds_k, N_D, N_S, N_D, 64, 32, 4, 3);
    {
      int r = t >> 3, c8 = (t & 7) * 8;      // v: 32 rows x 64 cols, 1 chunk/thread
      bf16x8 vv = *(const bf16x8*)&vp[(size_t)(j0 + r) * N_D + c8];
#pragma unroll
      for (int e = 0; e < 8; ++e)            // transpose v into LDS: vt[d][j]
        vt_s[(c8 + e) * G_LDS + r] = vv[e];
    }
    if (w == 0) __builtin_amdgcn_s_wait_tensorcnt(0);  // q (iter 0) + k landed
#else
    {
      int r = t >> 3, c8 = (t & 7) * 8;      // 32 rows x 64 cols, 1 chunk/thread
      *(bf16x8*)&k_s[r * 72 + c8] = *(const bf16x8*)&kp[(size_t)(j0 + r) * N_D + c8];
      bf16x8 vv = *(const bf16x8*)&vp[(size_t)(j0 + r) * N_D + c8];
#pragma unroll
      for (int e = 0; e < 8; ++e)            // transpose v into LDS: vt[d][j]
        vt_s[(c8 + e) * G_LDS + r] = vv[e];
    }
#endif
    __syncthreads();

    // s-tile: 16x16 per wave, K = 64 (two WMMA steps)
    v8f accs = (v8f)(0.0f);
#pragma unroll
    for (int d0 = 0; d0 < 64; d0 += 32) {
      v16bf a  = load_frag(q_s, wmS, 72, d0);
      v16bf bb = load_frag(k_s, wnS, 72, d0);
      accs = __builtin_amdgcn_wmma_f32_16x16x32_bf16(
          false, a, false, bb, (short)0, accs, false, false);
    }
    {
      int lj = wnS + cn, gj = j0 + lj;
#pragma unroll
      for (int p = 0; p < 8; ++p) {
        int li = wmS + rm0 + p, gi = i0 + li;
        float sv = accs[p] + pos_w[POS_C + gj - gi];
        sv = silu(sv);
        sv = (gj <= gi) ? sv : 0.0f;
        s_s[li * G_LDS + lj] = to_bf16(sv);
      }
    }
    __syncthreads();

    // y += s @ v : per wave 16x32, K = 32 (one step, two N subtiles)
    {
      v16bf a  = load_frag(s_s,  wmY,      G_LDS, 0);
      v16bf b0 = load_frag(vt_s, wnY,      G_LDS, 0);
      v16bf b1 = load_frag(vt_s, wnY + 16, G_LDS, 0);
      accy0 = __builtin_amdgcn_wmma_f32_16x16x32_bf16(
          false, a, false, b0, (short)0, accy0, false, false);
      accy1 = __builtin_amdgcn_wmma_f32_16x16x32_bf16(
          false, a, false, b1, (short)0, accy1, false, false);
    }
    __syncthreads();  // all waves done with k_s/vt_s/s_s before restaging
  }

  // y *= u, write [B,S,D]
#pragma unroll
  for (int p = 0; p < 8; ++p) {
    int gi = i0 + wmY + rm0 + p;
    size_t row = ((size_t)b * N_S + gi) * N_D + hh * N_HD;
    size_t i0d = row + (size_t)(wnY + cn);
    size_t i1d = row + (size_t)(wnY + 16 + cn);
    y[i0d] = accy0[p] * u[i0d];
    y[i1d] = accy1[p] * u[i1d];
  }
}

// ---- kernel 3/5: row layernorm (D = 1024), one block per row --------------

__global__ __launch_bounds__(256) void ln_bf16_kernel(
    const float* __restrict__ in, const float* __restrict__ g,
    const float* __restrict__ be, __bf16* __restrict__ out) {
  __shared__ float r1[256], r2[256];
  const size_t row = (size_t)blockIdx.x * N_D;
  const int t = threadIdx.x;
  float v[4], s1 = 0.f, s2 = 0.f;
#pragma unroll
  for (int i = 0; i < 4; ++i) {
    v[i] = in[row + t + 256 * i];
    s1 += v[i]; s2 += v[i] * v[i];
  }
  r1[t] = s1; r2[t] = s2;
  __syncthreads();
  for (int off = 128; off > 0; off >>= 1) {
    if (t < off) { r1[t] += r1[t + off]; r2[t] += r2[t + off]; }
    __syncthreads();
  }
  float mean = r1[0] * (1.0f / N_D);
  float var  = r2[0] * (1.0f / N_D) - mean * mean;
  float inv  = rsqrtf(var + 1e-5f);
#pragma unroll
  for (int i = 0; i < 4; ++i) {
    int c = t + 256 * i;
    out[row + c] = to_bf16((v[i] - mean) * inv * g[c] + be[c]);
  }
}

__global__ __launch_bounds__(256) void ln_f32_kernel(
    const float* __restrict__ in, const float* __restrict__ g,
    const float* __restrict__ be, float* __restrict__ out) {
  __shared__ float r1[256], r2[256];
  const size_t row = (size_t)blockIdx.x * N_D;
  const int t = threadIdx.x;
  float v[4], s1 = 0.f, s2 = 0.f;
#pragma unroll
  for (int i = 0; i < 4; ++i) {
    v[i] = in[row + t + 256 * i];
    s1 += v[i]; s2 += v[i] * v[i];
  }
  r1[t] = s1; r2[t] = s2;
  __syncthreads();
  for (int off = 128; off > 0; off >>= 1) {
    if (t < off) { r1[t] += r1[t + off]; r2[t] += r2[t + off]; }
    __syncthreads();
  }
  float mean = r1[0] * (1.0f / N_D);
  float var  = r2[0] * (1.0f / N_D) - mean * mean;
  float inv  = rsqrtf(var + 1e-5f);
#pragma unroll
  for (int i = 0; i < 4; ++i) {
    int c = t + 256 * i;
    out[row + c] = (v[i] - mean) * inv * g[c] + be[c];
  }
}

// ---- kernel 4: z = yn @ w2^T + b2 + x (residual) --------------------------
// M = 4096, N = 1024, K = 1024; same tiling as gemm1.

__global__ __launch_bounds__(256) void gemm2_kernel(
    const __bf16* __restrict__ A,     // ynb [4096,1024]
    const __bf16* __restrict__ Bw,    // w2b [1024,1024]
    const float*  __restrict__ bias2, // [1024]
    const float*  __restrict__ xres,  // x [4096,1024]
    float* __restrict__ z) {
#if HAVE_TDM
  __shared__ __bf16 As[2 * TILE_ELEMS];
  __shared__ __bf16 Bs[2 * TILE_ELEMS];
#else
  __shared__ __bf16 As[TILE_ELEMS];
  __shared__ __bf16 Bs[TILE_ELEMS];
#endif
  const int K = N_D;
  const int m0 = blockIdx.y * 128;
  const int n0 = blockIdx.x * 128;
  const int t = threadIdx.x;
  const int lane = t & 31, w = t >> 5;
  const int wm = (w >> 2) * 64;
  const int wn = (w & 3) * 32;

  v8f acc[4][2];
#pragma unroll
  for (int i = 0; i < 4; ++i)
#pragma unroll
    for (int j = 0; j < 2; ++j) acc[i][j] = (v8f)(0.0f);

#if HAVE_TDM
  const unsigned lds_a = (unsigned)(unsigned long long)(size_t)(void*)As;
  const unsigned lds_b = (unsigned)(unsigned long long)(size_t)(void*)Bs;
  if (w == 0) {
    tdm_load_2d(A  + (size_t)m0 * K, lds_a, K, N_R, K, 32, 128, 3, 3);
    tdm_load_2d(Bw + (size_t)n0 * K, lds_b, K, N_D, K, 32, 128, 3, 3);
  }
  for (int k0 = 0; k0 < K; k0 += 32) {
    const int cur = (k0 >> 5) & 1;
    const __bf16* Ac = As + cur * TILE_ELEMS;
    const __bf16* Bc = Bs + cur * TILE_ELEMS;
    if (w == 0) {
      if (k0 + 32 < K) {
        const int nb = cur ^ 1;
        tdm_load_2d(A  + (size_t)m0 * K + (k0 + 32), lds_a + nb * TILE_BYTES,
                    K, N_R, K, 32, 128, 3, 3);
        tdm_load_2d(Bw + (size_t)n0 * K + (k0 + 32), lds_b + nb * TILE_BYTES,
                    K, N_D, K, 32, 128, 3, 3);
        __builtin_amdgcn_s_wait_tensorcnt(2);
      } else {
        __builtin_amdgcn_s_wait_tensorcnt(0);
      }
    }
    __syncthreads();
    v16bf af[4], bfr[2];
#pragma unroll
    for (int i = 0; i < 4; ++i) af[i]  = load_frag(Ac, wm + i * 16, G_LDS, 0);
#pragma unroll
    for (int j = 0; j < 2; ++j) bfr[j] = load_frag(Bc, wn + j * 16, G_LDS, 0);
#pragma unroll
    for (int i = 0; i < 4; ++i)
#pragma unroll
      for (int j = 0; j < 2; ++j)
        acc[i][j] = __builtin_amdgcn_wmma_f32_16x16x32_bf16(
            false, af[i], false, bfr[j], (short)0, acc[i][j], false, false);
    __syncthreads();
  }
#else
  for (int k0 = 0; k0 < K; k0 += 32) {
#pragma unroll
    for (int i = 0; i < 2; ++i) {
      int c = t + 256 * i;
      int r = c >> 2, c8 = (c & 3) * 8;
      *(bf16x8*)&As[r * G_LDS + c8] = *(const bf16x8*)&A [(size_t)(m0 + r) * K + k0 + c8];
      *(bf16x8*)&Bs[r * G_LDS + c8] = *(const bf16x8*)&Bw[(size_t)(n0 + r) * K + k0 + c8];
    }
    __syncthreads();
    v16bf af[4], bfr[2];
#pragma unroll
    for (int i = 0; i < 4; ++i) af[i]  = load_frag(As, wm + i * 16, G_LDS, 0);
#pragma unroll
    for (int j = 0; j < 2; ++j) bfr[j] = load_frag(Bs, wn + j * 16, G_LDS, 0);
#pragma unroll
    for (int i = 0; i < 4; ++i)
#pragma unroll
      for (int j = 0; j < 2; ++j)
        acc[i][j] = __builtin_amdgcn_wmma_f32_16x16x32_bf16(
            false, af[i], false, bfr[j], (short)0, acc[i][j], false, false);
    __syncthreads();
  }
#endif

  const int cn = lane & 15, rm0 = (lane >> 4) * 8;
#pragma unroll
  for (int i = 0; i < 4; ++i)
#pragma unroll
    for (int j = 0; j < 2; ++j) {
      int n = n0 + wn + j * 16 + cn;
      float bv = bias2[n];
#pragma unroll
      for (int p = 0; p < 8; ++p) {
        int m = m0 + wm + i * 16 + rm0 + p;
        size_t idx = (size_t)m * N_D + n;
        z[idx] = acc[i][j][p] + bv + xres[idx];
      }
    }
}

// ---- launch ---------------------------------------------------------------

extern "C" void kernel_launch(void* const* d_in, const int* in_sizes, int n_in,
                              void* d_out, int out_size, void* d_ws, size_t ws_size,
                              hipStream_t stream) {
  const float* x     = (const float*)d_in[0];
  // d_in[1] = mask (int32 causal): semantics hard-coded (j <= i keep)
  const float* w1    = (const float*)d_in[2];
  const float* b1    = (const float*)d_in[3];
  const float* w2    = (const float*)d_in[4];
  const float* b2    = (const float*)d_in[5];
  const float* g1    = (const float*)d_in[6];
  const float* be1   = (const float*)d_in[7];
  const float* g2    = (const float*)d_in[8];
  const float* be2   = (const float*)d_in[9];
  const float* pos_w = (const float*)d_in[10];
  float* out = (float*)d_out;

  char* ws = (char*)d_ws;
  size_t off = 0;
  auto nxt = [&](size_t bytes) -> void* {
    void* p = ws + off;
    off += (bytes + 255) & ~(size_t)255;
    return p;
  };
  const size_t RD = (size_t)N_R * N_D;          // 4M elements
  __bf16* xb  = (__bf16*)nxt(RD * 2);           // x  in bf16
  __bf16* w1b = (__bf16*)nxt((size_t)N_4D * N_D * 2);
  __bf16* w2b = (__bf16*)nxt((size_t)N_D * N_D * 2);
  float*  u   = (float* )nxt(RD * 4);
  __bf16* qb  = (__bf16*)nxt(RD * 2);
  __bf16* kb  = (__bf16*)nxt(RD * 2);
  __bf16* vb  = (__bf16*)nxt(RD * 2);
  float*  y   = (float* )nxt(RD * 4);
  __bf16* ynb = (__bf16*)nxt(RD * 2);
  float*  z   = (float* )nxt(RD * 4);
  (void)ws_size; (void)in_sizes; (void)n_in; (void)out_size;

  // 0) precision conversion
  cvt_bf16_kernel<<<(int)(RD / 256), 256, 0, stream>>>(x,  xb,  (int)RD);
  cvt_bf16_kernel<<<(int)(RD / 256), 256, 0, stream>>>(w1, w1b, (int)RD);
  cvt_bf16_kernel<<<(int)((size_t)N_D * N_D / 256), 256, 0, stream>>>(w2, w2b, N_D * N_D);

  // 1) h = silu(x w1^T + b1) -> u, q, k, v
  gemm1_kernel<<<dim3(N_4D / 128, N_R / 128), 256, 0, stream>>>(xb, w1b, b1, u, qb, kb, vb);

  // 2) fused causal silu-attention, y = (att v) * u
  attn_kernel<<<dim3(N_S / 64, N_B * N_H), 256, 0, stream>>>(qb, kb, vb, u, pos_w, y);

  // 3) layernorm1 -> bf16
  ln_bf16_kernel<<<N_R, 256, 0, stream>>>(y, g1, be1, ynb);

  // 4) z = yn w2^T + b2 + x
  gemm2_kernel<<<dim3(N_D / 128, N_R / 128), 256, 0, stream>>>(ynb, w2b, b2, x, z);

  // 5) layernorm2 -> out
  ln_f32_kernel<<<N_R, 256, 0, stream>>>(z, g2, be2, out);
}
